// SeparateEncoder_7834020348590
// MI455X (gfx1250) — matrix operand order repaired and checked by reading, not compiled, not run on hardware
//
#include <hip/hip_runtime.h>
#include <hip/hip_bf16.h>

#define BN_EPS 1e-5f

typedef __attribute__((ext_vector_type(16))) __bf16 v16bf;
typedef __attribute__((ext_vector_type(4)))  __bf16 v4bf;
typedef __attribute__((ext_vector_type(8)))  float  v8f;

// ---------------------------------------------------------------- utilities

__global__ void fill_f32_kernel(float* __restrict__ p, size_t n, float v) {
    size_t i = (size_t)blockIdx.x * blockDim.x + threadIdx.x;
    size_t stride = (size_t)gridDim.x * blockDim.x;
    for (; i < n; i += stride) p[i] = v;
}

__global__ void fill_u32_kernel(unsigned* __restrict__ p, size_t n, unsigned v) {
    size_t i = (size_t)blockIdx.x * blockDim.x + threadIdx.x;
    size_t stride = (size_t)gridDim.x * blockDim.x;
    for (; i < n; i += stride) p[i] = v;
}

// degree count with native u32 atomics (self loop handled by init=1).
// int64 indices: only low dword is meaningful (N < 2^31).
__global__ void degree_kernel(const int* __restrict__ dst32,
                              unsigned* __restrict__ deg, int nEdges) {
    int i = blockIdx.x * blockDim.x + threadIdx.x;
    int stride = gridDim.x * blockDim.x;
    for (; i < nEdges; i += stride)
        atomicAdd(&deg[dst32[2 * i]], 1u);
}

__global__ void rsqrt_deg_kernel(const unsigned* __restrict__ deg,
                                 float* __restrict__ dinv, int n) {
    int i = blockIdx.x * blockDim.x + threadIdx.x;
    int stride = gridDim.x * blockDim.x;
    for (; i < n; i += stride) dinv[i] = rsqrtf((float)deg[i]);
}

// ---------------------------------------------------------------- WMMA GEMM
// out[N x DOUT] = h[N x DIN] @ W[DIN x DOUT], bf16 A/B fragments, f32 acc.
// 8 waves / 256-thread block; each wave owns a 16-row tile; K padded to KPAD.
// W is staged PRE-SWIZZLED into the B-fragment layout (2x ds_load_b128, no
// packing). A fragments are hoisted into registers once and reused across
// all column tiles. Epilogue goes through LDS for coalesced b128 stores.

template<int DIN, int DOUT>
__global__ void gemm_wmma_kernel(const float* __restrict__ h,
                                 const float* __restrict__ W,
                                 float* __restrict__ out, int n) {
    constexpr int KPAD  = (DIN + 31) & ~31;
    constexpr int NW    = 8;
    constexpr int NTILE = DOUT / 16;
    constexpr int KTILE = KPAD / 32;
    static_assert(DIN % 4 == 0 && DOUT % 16 == 0, "dims");

    // B fragments: per (col-tile, k-chunk), 32 lanes x 16 bf16 contiguous.
    __shared__ __attribute__((aligned(32))) __bf16 Wswz[NTILE][KTILE][32][16];
    __shared__ __bf16 Al[NW][16][KPAD];
    __shared__ float  Cl[NW][16][20];        // pad 16->20: conflict-free, 16B rows

    const int tid  = threadIdx.x;
    const int wave = tid >> 5;
    const int lane = tid & 31;
    const int rowbase = blockIdx.x * (NW * 16) + wave * 16;

    // B layout used here, contiguous k (32x16 bf16, wave32): lanes 0-15 col N=lane hold
    // K 0..15; lanes 16-31 hold K 16..31. Stage W directly in that shape.
    for (int i = tid; i < NTILE * KTILE * 32 * 4; i += blockDim.x) {
        int jq = i & 3;
        int ln = (i >> 2) & 31;
        int kb = (i >> 7) % KTILE;
        int nt = (i >> 7) / KTILE;
        int c     = nt * 16 + (ln & 15);
        int kbase = kb * 32 + ((ln >> 4) ? 16 : 0) + jq * 4;
        v4bf p;
#pragma unroll
        for (int jj = 0; jj < 4; ++jj) {
            int k = kbase + jj;
            float v = (k < DIN) ? W[k * DOUT + c] : 0.0f;
            p[jj] = (__bf16)v;
        }
        *(v4bf*)&Wswz[nt][kb][ln][jq * 4] = p;
    }
    // stage this wave's 16 A rows (zero-pad K >= DIN, rows >= n)
    constexpr int K4 = KPAD / 4;
    for (int i = lane; i < 16 * K4; i += 32) {
        int r = i / K4, k = (i % K4) * 4;
        int row = rowbase + r;
        float4 v = make_float4(0.f, 0.f, 0.f, 0.f);
        if (row < n && k < DIN) v = *(const float4*)(h + (size_t)row * DIN + k);
        v4bf p; p[0] = (__bf16)v.x; p[1] = (__bf16)v.y; p[2] = (__bf16)v.z; p[3] = (__bf16)v.w;
        *(v4bf*)&Al[wave][r][k] = p;
    }
    __syncthreads();

    // A 16x32 bf16 layout: lanes 0-15 row M=lane hold K 0..15;
    // lanes 16-31 hold K 16..31.
    const int mrow  = lane & 15;
    const int ncol  = lane & 15;
    const int hiA   = (lane >> 4) ? 16 : 0;
    const int mbase = (lane >> 4) ? 8 : 0;

    // Hoist A fragments into registers: invariant across column tiles.
    v16bf afrag[KTILE];
#pragma unroll
    for (int kb = 0; kb < KTILE; ++kb) {
        const int K0 = kb * 32;
#pragma unroll
        for (int j = 0; j < 8; ++j) {
            afrag[kb][j]     = Al[wave][mrow][K0 + hiA + j];
            afrag[kb][8 + j] = Al[wave][mrow][K0 + hiA + 8 + j];
        }
    }

    for (int nt = 0; nt < NTILE; ++nt) {
        v8f acc = {};
#pragma unroll
        for (int kb = 0; kb < KTILE; ++kb) {
            v16bf b = *(const v16bf*)&Wswz[nt][kb][lane][0];
            acc = __builtin_amdgcn_wmma_f32_16x16x32_bf16(
                false, afrag[kb], false, b, (short)0, acc, false, false);
        }
        // C/D layout: VGPR r, lanes 0-15 -> M=r, lanes 16-31 -> M=8+r, N=lane&15.
        // Cl slab is per-wave; LDS ops from one wave complete in order, so a
        // compiler scheduling barrier suffices (no block barrier).
#pragma unroll
        for (int r = 0; r < 8; ++r)
            Cl[wave][mbase + r][ncol] = acc[r];
        __builtin_amdgcn_wave_barrier();
        // coalesced b128 stores: 64 float4s per tile, 2 per lane
#pragma unroll
        for (int u = 0; u < 2; ++u) {
            int fi = lane * 2 + u;          // 0..63
            int r  = fi >> 2;               // row in tile
            int cq = fi & 3;                // float4 within row
            int row = rowbase + r;
            if (row < n) {
                float4 vv = *(const float4*)&Cl[wave][r][cq * 4];
                *(float4*)(out + (size_t)row * DOUT + nt * 16 + cq * 4) = vv;
            }
        }
        __builtin_amdgcn_wave_barrier();
    }
}

// ---------------------------------------------------------------- scatter
// agg[dst] += m[src] * dinv[src]*dinv[dst]; edges [0,E) from list, [E,E+N)
// are self loops. Thread owns a fixed quad q, strides over edges: no div/mod
// in the hot loop. float4 gather + 4 native f32 atomics per edge-quad.
// Edge indices read as narrow i32 (low dword of the int64).

template<int DOUT>
__global__ void scatter_kernel(const float* __restrict__ m,
                               const int* __restrict__ src32,
                               const int* __restrict__ dst32,
                               const float* __restrict__ dinv,
                               float* __restrict__ agg,
                               int nEdges, int nNodes) {
    constexpr int QP = DOUT / 4;            // blockDim is a multiple of QP
    const int gt = blockIdx.x * blockDim.x + threadIdx.x;
    const int q = gt % QP;
    const int estride = (gridDim.x * blockDim.x) / QP;
    const int total = nEdges + nNodes;
    for (int e = gt / QP; e < total; e += estride) {
        int s, d; float nrm;
        if (e < nEdges) {
            __builtin_prefetch(src32 + 2 * (e + 4 * estride), 0, 1);
            __builtin_prefetch(dst32 + 2 * (e + 4 * estride), 0, 1);
            s = src32[2 * e];
            d = dst32[2 * e];
            nrm = dinv[s] * dinv[d];
        } else {
            s = d = e - nEdges;
            float di = dinv[s];
            nrm = di * di;
        }
        const float4 v = *(const float4*)(m + (size_t)s * DOUT + q * 4);
        float* o = agg + (size_t)d * DOUT + q * 4;
        unsafeAtomicAdd(o + 0, v.x * nrm);
        unsafeAtomicAdd(o + 1, v.y * nrm);
        unsafeAtomicAdd(o + 2, v.z * nrm);
        unsafeAtomicAdd(o + 3, v.w * nrm);
    }
}

// ---------------------------------------------------------------- batchnorm

template<int D>
__global__ void bnstats_kernel(const float* __restrict__ x,
                               const float* __restrict__ b,
                               float* __restrict__ stats, int n) {
    // blockDim.x == 4*D; thread owns (col, row-group)
    const int col = threadIdx.x % D;
    const int rg  = threadIdx.x / D;
    const float bias = b[col];
    float s = 0.0f, s2 = 0.0f;
    for (int r = blockIdx.x * 4 + rg; r < n; r += gridDim.x * 4) {
        float v = x[(size_t)r * D + col] + bias;
        s += v; s2 += v * v;
    }
    unsafeAtomicAdd(&stats[col], s);
    unsafeAtomicAdd(&stats[D + col], s2);
}

template<int D, bool RELU>
__global__ void bnapply_kernel(const float* __restrict__ x,
                               const float* __restrict__ b,
                               const float* __restrict__ g,
                               const float* __restrict__ beta,
                               const float* __restrict__ stats,
                               float* __restrict__ out, int n) {
    const float invN = 1.0f / (float)n;
    size_t total = (size_t)n * D;
    size_t i = (size_t)blockIdx.x * blockDim.x + threadIdx.x;
    size_t stride = (size_t)gridDim.x * blockDim.x;
    for (; i < total; i += stride) {
        int col = (int)(i % D);
        float mu  = stats[col] * invN;
        float var = stats[D + col] * invN - mu * mu;
        float sc  = g[col] * rsqrtf(var + BN_EPS);
        float v = (x[i] + b[col] - mu) * sc + beta[col];
        if (RELU) v = fmaxf(v, 0.0f);
        out[i] = v;
    }
}

// ---------------------------------------------------------------- layer driver

template<int DIN, int DOUT, bool RELU>
static void run_layer(const float* hin, const float* W, const float* b,
                      const float* g, const float* beta,
                      const int* src32, const int* dst32,
                      const float* dinv, float* mbuf, float* agg, float* stats,
                      float* hout, int n, int E, hipStream_t s) {
    constexpr int SBLK = (DOUT / 4) * 16;   // 256 / 192 / 128, multiple of QP
    gemm_wmma_kernel<DIN, DOUT><<<(n + 127) / 128, 256, 0, s>>>(hin, W, mbuf, n);
    fill_f32_kernel<<<2048, 256, 0, s>>>(agg, (size_t)n * DOUT, 0.0f);
    fill_f32_kernel<<<1, 128, 0, s>>>(stats, 2 * DOUT, 0.0f);
    scatter_kernel<DOUT><<<4096, SBLK, 0, s>>>(mbuf, src32, dst32, dinv, agg, E, n);
    bnstats_kernel<DOUT><<<1024, 4 * DOUT, 0, s>>>(agg, b, stats, n);
    bnapply_kernel<DOUT, RELU><<<2048, 256, 0, s>>>(agg, b, g, beta, stats, hout, n);
}

// ---------------------------------------------------------------- entry

extern "C" void kernel_launch(void* const* d_in, const int* in_sizes, int n_in,
                              void* d_out, int out_size, void* d_ws, size_t ws_size,
                              hipStream_t stream) {
    const float* x  = (const float*)d_in[0];
    const int*   ei = (const int*)d_in[1];     // int64 [2, E]; use low dwords
    const float* W1 = (const float*)d_in[2];
    const float* b1 = (const float*)d_in[3];
    const float* g1 = (const float*)d_in[4];
    const float* be1 = (const float*)d_in[5];
    const float* W2 = (const float*)d_in[6];
    const float* b2 = (const float*)d_in[7];
    const float* g2 = (const float*)d_in[8];
    const float* be2 = (const float*)d_in[9];
    const float* W3 = (const float*)d_in[10];
    const float* b3 = (const float*)d_in[11];
    const float* g3 = (const float*)d_in[12];
    const float* be3 = (const float*)d_in[13];
    float* out = (float*)d_out;

    const int N = in_sizes[0] / 20;     // 100000
    const int E = in_sizes[1] / 2;      // 1600000
    const int* src32 = ei;              // int64 row 0 (low dwords at 2*e)
    const int* dst32 = ei + 2 * E;      // int64 row 1

    // workspace carve-out (256B aligned)
    char* ws = (char*)d_ws;
    size_t off = 0;
    auto carve = [&](size_t bytes) -> void* {
        void* p = (void*)(ws + off);
        off = (off + bytes + 255) & ~(size_t)255;
        return p;
    };
    float*    hbuf  = (float*)carve((size_t)N * 64 * sizeof(float));
    float*    mbuf  = (float*)carve((size_t)N * 64 * sizeof(float));
    float*    agg   = (float*)carve((size_t)N * 64 * sizeof(float));
    unsigned* deg   = (unsigned*)carve((size_t)N * sizeof(unsigned));
    float*    dinv  = (float*)carve((size_t)N * sizeof(float));
    float*    stats = (float*)carve(128 * sizeof(float));
    (void)ws_size; (void)n_in; (void)out_size;

    // degree (init 1 = self loop) -> dinv = rsqrt(deg)
    fill_u32_kernel<<<1024, 256, 0, stream>>>(deg, (size_t)N, 1u);
    degree_kernel<<<4096, 256, 0, stream>>>(dst32, deg, E);
    rsqrt_deg_kernel<<<1024, 256, 0, stream>>>(deg, dinv, N);

    // layer 1: 20 -> 64, BN + ReLU
    run_layer<20, 64, true >(x,    W1, b1, g1, be1, src32, dst32, dinv, mbuf, agg, stats, hbuf, N, E, stream);
    // layer 2: 64 -> 48, BN + ReLU
    run_layer<64, 48, true >(hbuf, W2, b2, g2, be2, src32, dst32, dinv, mbuf, agg, stats, hbuf, N, E, stream);
    // layer 3: 48 -> 32, BN only
    run_layer<48, 32, false>(hbuf, W3, b3, g3, be3, src32, dst32, dinv, mbuf, agg, stats, out, N, E, stream);
}
